// TrajectoryVAE_19808389169824
// MI455X (gfx1250) — matrix operand order repaired and checked by reading, not compiled, not run on hardware
//
#include <hip/hip_runtime.h>

// ---------------------------------------------------------------------------
// TrajectoryVAE on gfx1250 (MI455X): WMMA f16->f32 LSTM scans + GEMMs.
// B=256, T=250, V=170, E=128, L=32, Hd=256.
// - Weights pre-packed into WMMA B-fragment order (32B/lane contiguous loads)
// - Fwd/bwd LSTM directions fused into one launch (gridDim.y = 2)
// - Broadcast-z input projection hoisted out of the decoder-1 time loop
// ---------------------------------------------------------------------------

#define BB 256
#define TT 250
#define VV 170
#define EE 128
#define LL 32
#define HD 256

typedef __attribute__((ext_vector_type(16))) _Float16 v16h;
typedef __attribute__((ext_vector_type(8)))  _Float16 v8h;
typedef __attribute__((ext_vector_type(8)))  float    v8f;

// ---- WMMA helpers ----------------------------------------------------------

__device__ __forceinline__ v8f wmma16(v16h a, v16h b, v8f c) {
  return __builtin_amdgcn_wmma_f32_16x16x32_f16(
      /*neg_a=*/false, a, /*neg_b=*/false, b,
      /*c_mod=*/(short)0, c, /*reuse_a=*/false, /*reuse_b=*/false);
}

// Load 16x32 f16 A-tile from row-major src (row stride lda, elements).
// Per ISA 7.12.2: lane l<16 -> M=l, K base 8*hi (+16 for upper 8 halves).
// Two 16B vector loads per lane.
__device__ __forceinline__ v16h load_a16(const _Float16* __restrict__ src,
                                         int lda, int lane) {
  const int l  = lane & 15;
  const int hi = lane >> 4;
  const _Float16* row = src + (long)l * lda + 8 * hi;
  v8h lo = *(const v8h*)(row);
  v8h hh = *(const v8h*)(row + 16);
  return __builtin_shufflevector(lo, hh, 0, 1, 2, 3, 4, 5, 6, 7,
                                 8, 9, 10, 11, 12, 13, 14, 15);
}

// Load one pre-packed 32x16 B-tile: 32B contiguous per lane.
__device__ __forceinline__ v16h load_bp(const _Float16* __restrict__ base,
                                        long tile, int lane) {
  return *(const v16h*)(base + (tile * 32 + lane) * 16);
}

__device__ __forceinline__ float sigf(float x) {
  return 1.f / (1.f + __expf(-x));
}

// ---- weight pre-pack: row-major f32 [K,N] -> B-fragment-ordered f16 --------
// dst[((ktile*NT + ntile)*32 + lane)*16 + j] = src[(ktile*32+8*hi+j+(j>=8?8:0))*N + ntile*16+n]
// where n = lane&15, hi = lane>>4. Pads cols N..Npad-1 with zeros.

__global__ void pack_b_f16(const float* __restrict__ src,
                           _Float16* __restrict__ dst, int K, int N, int Npad) {
  long i = (long)blockIdx.x * 256 + threadIdx.x;
  long total = (long)K * Npad;
  if (i >= total) return;
  const int j    = (int)(i & 15);
  const int lane = (int)((i >> 4) & 31);
  const long tile = i >> 9;
  const int NT   = Npad >> 4;
  const int ntile = (int)(tile % NT);
  const int ktile = (int)(tile / NT);
  const int k   = ktile * 32 + 8 * (lane >> 4) + j + ((j >= 8) ? 8 : 0);
  const int col = ntile * 16 + (lane & 15);
  float v = (col < N) ? src[(long)k * N + col] : 0.f;
  dst[i] = (_Float16)v;
}

// ---- embedding gather -> f16 -----------------------------------------------

__global__ void embed_kernel(const int* __restrict__ x,
                             const float* __restrict__ emb,
                             _Float16* __restrict__ e, long n) {
  long i = (long)blockIdx.x * 256 + threadIdx.x;
  if (i < n) {
    long r = i >> 7;          // token position (b*T+t)
    int  c = (int)(i & 127);  // embed column
    int tok = x[r];
    e[i] = (_Float16)emb[(long)tok * EE + c];
  }
}

// ---- fused bidirectional LSTM scan -----------------------------------------
// gridDim = (B/16, 2): blockIdx.y selects direction (0=fwd, 1=bwd/time-rev).
// One block = 16 batch rows, 8 waves. Each wave owns H/8 columns of EVERY gate
// (i,f,g,o strided by H): gate math happens on WMMA fragments in registers and
// cell state stays in registers. Only the 16xH f16 h-tile touches LDS.
// zg = x_t @ Wi + h @ Wh + bias ; i,f,g,o LSTM update.
// XBCAST: input is time-invariant (xst==0) -> x@Wi hoisted out of the t-loop.

template <int DIN, int H, bool XBCAST, bool OUT_F16>
__global__ void __launch_bounds__(256)
lstm_scan(const _Float16* __restrict__ xseq, long xsb, long xst,
          const _Float16* __restrict__ Wi_f, const _Float16* __restrict__ Wh_f,
          const float* __restrict__ bias_f,
          const _Float16* __restrict__ Wi_b, const _Float16* __restrict__ Wh_b,
          const float* __restrict__ bias_b,
          void* __restrict__ outp, long osb, long ost, int T) {
  constexpr int G   = 4 * H;
  constexpr int GT  = G / 16;  // total 16-col tiles across gates
  constexpr int CW  = H / 8;   // gate columns per wave
  constexpr int NT2 = CW / 16; // 16-col tiles per gate per wave

  __shared__ _Float16 s_h[16 * H];

  const int dir = blockIdx.y;
  const _Float16* __restrict__ Wi = dir ? Wi_b : Wi_f;
  const _Float16* __restrict__ Wh = dir ? Wh_b : Wh_f;
  const float*    __restrict__ bias = dir ? bias_b : bias_f;
  const int ocol0 = dir * H;

  const int tid  = threadIdx.x;
  const int lane = tid & 31;
  const int wave = tid >> 5;
  const int hi   = lane >> 4;
  const int n    = lane & 15;
  const long mrow0 = (long)blockIdx.x * 16;

  for (int i = tid; i < 16 * H; i += 256) s_h[i] = (_Float16)0.f;

  // Bias in fragment layout (col-only), cell state in fragment layout.
  float breg[4][NT2];
#pragma unroll
  for (int g = 0; g < 4; ++g)
#pragma unroll
    for (int s = 0; s < NT2; ++s)
      breg[g][s] = bias[g * H + wave * CW + s * 16 + n];

  float creg[NT2][8];
#pragma unroll
  for (int s = 0; s < NT2; ++s)
#pragma unroll
    for (int r = 0; r < 8; ++r) creg[s][r] = 0.f;

  // Time-invariant input projection (decoder-1 broadcast z).
  v8f accx[4][NT2];
  if (XBCAST) {
#pragma unroll
    for (int g = 0; g < 4; ++g)
#pragma unroll
      for (int s = 0; s < NT2; ++s) accx[g][s] = {};
    const _Float16* xt = xseq + mrow0 * xsb;
    for (int k = 0; k < DIN; k += 32) {
      v16h a = load_a16(xt + k, (int)xsb, lane);
      const long kt = (long)(k >> 5) * GT;
#pragma unroll
      for (int g = 0; g < 4; ++g)
#pragma unroll
        for (int s = 0; s < NT2; ++s) {
          const int ntile = g * (H / 16) + wave * NT2 + s;
          v16h b = load_bp(Wi, kt + ntile, lane);
          accx[g][s] = wmma16(a, b, accx[g][s]);
        }
    }
  }

  __syncthreads();

  for (int t = 0; t < T; ++t) {
    const int te = dir ? (T - 1 - t) : t;

    v8f acc[4][NT2];
    if (XBCAST) {
#pragma unroll
      for (int g = 0; g < 4; ++g)
#pragma unroll
        for (int s = 0; s < NT2; ++s) acc[g][s] = accx[g][s];
    } else {
#pragma unroll
      for (int g = 0; g < 4; ++g)
#pragma unroll
        for (int s = 0; s < NT2; ++s) acc[g][s] = {};
      // x_t @ Wi  (A: global b128 loads; B: packed fragment loads from L2)
      const _Float16* xt = xseq + mrow0 * xsb + (long)te * xst;
      for (int k = 0; k < DIN; k += 32) {
        v16h a = load_a16(xt + k, (int)xsb, lane);
        const long kt = (long)(k >> 5) * GT;
#pragma unroll
        for (int g = 0; g < 4; ++g)
#pragma unroll
          for (int s = 0; s < NT2; ++s) {
            const int ntile = g * (H / 16) + wave * NT2 + s;
            v16h b = load_bp(Wi, kt + ntile, lane);
            acc[g][s] = wmma16(a, b, acc[g][s]);
          }
      }
    }
    // h @ Wh  (A: LDS ds_load_b128)
    for (int k = 0; k < H; k += 32) {
      v16h a = load_a16(s_h + k, H, lane);
      const long kt = (long)(k >> 5) * GT;
#pragma unroll
      for (int g = 0; g < 4; ++g)
#pragma unroll
        for (int s = 0; s < NT2; ++s) {
          const int ntile = g * (H / 16) + wave * NT2 + s;
          v16h b = load_bp(Wh, kt + ntile, lane);
          acc[g][s] = wmma16(a, b, acc[g][s]);
        }
    }
    __syncthreads();  // all waves done reading s_h

    // Gate nonlinearities directly on fragments.
#pragma unroll
    for (int s = 0; s < NT2; ++s) {
#pragma unroll
      for (int r = 0; r < 8; ++r) {
        const int row = r + 8 * hi;
        const int col = wave * CW + s * 16 + n;
        float zi = acc[0][s][r] + breg[0][s];
        float zf = acc[1][s][r] + breg[1][s];
        float zg = acc[2][s][r] + breg[2][s];
        float zo = acc[3][s][r] + breg[3][s];
        float c = sigf(zf) * creg[s][r] + sigf(zi) * tanhf(zg);
        float h = sigf(zo) * tanhf(c);
        creg[s][r] = c;
        s_h[row * H + col] = (_Float16)h;
        long oidx = (mrow0 + row) * osb + (long)te * ost + ocol0 + col;
        if (OUT_F16) ((_Float16*)outp)[oidx] = (_Float16)h;
        else         ((float*)outp)[oidx] = h;
      }
    }
    __syncthreads();  // s_h updated for next step
  }
}

// ---- masked mean-pool + mean/logvar + reparameterize -----------------------

__global__ void __launch_bounds__(256)
pool_kernel(const int* __restrict__ x, const float* __restrict__ hs,
            const float* __restrict__ Wm, const float* __restrict__ bm,
            const float* __restrict__ Wv, const float* __restrict__ bv,
            const float* __restrict__ eps, float* __restrict__ mean_out,
            float* __restrict__ logvar_out, _Float16* __restrict__ z16,
            int T) {
  __shared__ float s_pool[2 * EE];
  const int b = blockIdx.x;
  const int j = threadIdx.x;  // 0..255 over 2E features

  float acc = 0.f, cnt = 0.f;
  for (int t = 0; t < T; ++t) {
    float m = (x[b * T + t] > 0) ? 1.f : 0.f;
    cnt += m;
    acc += m * hs[((long)b * T + t) * (2 * EE) + j];
  }
  s_pool[j] = acc / fmaxf(cnt, 1.f);
  __syncthreads();

  if (j < LL) {
    float zm = 0.f, zv = 0.f;
    for (int k = 0; k < 2 * EE; ++k) {
      float p = s_pool[k];
      zm += p * Wm[k * LL + j];
      zv += p * Wv[k * LL + j];
    }
    float mean = tanhf(zm + bm[j]) * 6.f;
    float lv = zv + bv[j];
    lv = fminf(fmaxf(lv, -10.f), 4.f);
    float z = mean + eps[b * LL + j] * __expf(0.5f * lv);
    mean_out[b * LL + j]   = mean;
    logvar_out[b * LL + j] = lv;
    z16[b * LL + j] = (_Float16)z;
  }
}

// ---- final projection GEMM: [M,K] f16 @ packed[K,Npad] f16 + bias -> f32 ---

__global__ void __launch_bounds__(256)
out_gemm(const _Float16* __restrict__ A, const _Float16* __restrict__ Bp,
         const float* __restrict__ bias, float* __restrict__ C, int M, int N,
         int Npad, int K) {
  const int lane = threadIdx.x & 31;
  const int wave = threadIdx.x >> 5;
  const int hi = lane >> 4, n = lane & 15;
  const long m0 = (long)blockIdx.x * 16;
  const int NT = Npad >> 4;
  const int nt = blockIdx.y * 8 + wave;
  if (nt >= NT) return;  // wave-uniform: EXEC stays all-ones for WMMA

  v8f acc = {};
  for (int k = 0; k < K; k += 32) {
    v16h a = load_a16(A + m0 * K + k, K, lane);
    v16h b = load_bp(Bp, (long)(k >> 5) * NT + nt, lane);
    acc = wmma16(a, b, acc);
  }
  const int col = nt * 16 + n;
  if (col < N) {
#pragma unroll
    for (int r = 0; r < 8; ++r) {
      long row = m0 + r + 8 * hi;
      C[row * N + col] = acc[r] + bias[col];
    }
  }
}

// ---------------------------------------------------------------------------

extern "C" void kernel_launch(void* const* d_in, const int* in_sizes, int n_in,
                              void* d_out, int out_size, void* d_ws,
                              size_t ws_size, hipStream_t stream) {
  (void)in_sizes; (void)n_in; (void)out_size; (void)ws_size;

  // ---- inputs (setup_inputs dict order) ----
  const int*   x        = (const int*)  d_in[0];
  const float* eps      = (const float*)d_in[1];
  const float* embed    = (const float*)d_in[2];
  const float* enc_b_f  = (const float*)d_in[5];
  const float* enc_b_b  = (const float*)d_in[8];
  const float* W_mean   = (const float*)d_in[9];
  const float* b_mean   = (const float*)d_in[10];
  const float* W_logvar = (const float*)d_in[11];
  const float* b_logvar = (const float*)d_in[12];
  const float* d1_b_f   = (const float*)d_in[15];
  const float* d1_b_b   = (const float*)d_in[18];
  const float* d2_b_f   = (const float*)d_in[21];
  const float* d2_b_b   = (const float*)d_in[24];
  const float* b_out    = (const float*)d_in[26];

  // ---- workspace carve-out ----
  char* p = (char*)d_ws;
  auto alloc = [&](size_t bytes) {
    void* r = (void*)p;
    p += (bytes + 255) & ~(size_t)255;
    return r;
  };
  const int VPAD = 176;  // 170 padded to 16
  _Float16* encWi_f = (_Float16*)alloc(EE * 4 * EE * 2);
  _Float16* encWh_f = (_Float16*)alloc(EE * 4 * EE * 2);
  _Float16* encWi_b = (_Float16*)alloc(EE * 4 * EE * 2);
  _Float16* encWh_b = (_Float16*)alloc(EE * 4 * EE * 2);
  _Float16* d1Wi_f  = (_Float16*)alloc(LL * 4 * HD * 2);
  _Float16* d1Wh_f  = (_Float16*)alloc(HD * 4 * HD * 2);
  _Float16* d1Wi_b  = (_Float16*)alloc(LL * 4 * HD * 2);
  _Float16* d1Wh_b  = (_Float16*)alloc(HD * 4 * HD * 2);
  _Float16* d2Wi_f  = (_Float16*)alloc(2 * HD * 4 * HD * 2);
  _Float16* d2Wh_f  = (_Float16*)alloc(HD * 4 * HD * 2);
  _Float16* d2Wi_b  = (_Float16*)alloc(2 * HD * 4 * HD * 2);
  _Float16* d2Wh_b  = (_Float16*)alloc(HD * 4 * HD * 2);
  _Float16* Wout16  = (_Float16*)alloc(2 * HD * VPAD * 2);
  _Float16* e16     = (_Float16*)alloc((size_t)BB * TT * EE * 2);
  _Float16* z16     = (_Float16*)alloc((size_t)BB * LL * 2);
  float*    hs_enc  = (float*)   alloc((size_t)BB * TT * 2 * EE * 4);
  _Float16* d1o     = (_Float16*)alloc((size_t)BB * TT * 2 * HD * 2);
  _Float16* d2o     = (_Float16*)alloc((size_t)BB * TT * 2 * HD * 2);

  // ---- outputs ----
  float* logits     = (float*)d_out;
  float* mean_out   = logits + (size_t)BB * TT * VV;
  float* logvar_out = mean_out + (size_t)BB * LL;

  // ---- 1) pack weights f32 -> fragment-ordered f16 ----
  struct Job { const float* src; _Float16* dst; int K, N, Npad; };
  const Job jobs[] = {
      {(const float*)d_in[3],  encWi_f, EE,     4 * EE, 4 * EE},
      {(const float*)d_in[4],  encWh_f, EE,     4 * EE, 4 * EE},
      {(const float*)d_in[6],  encWi_b, EE,     4 * EE, 4 * EE},
      {(const float*)d_in[7],  encWh_b, EE,     4 * EE, 4 * EE},
      {(const float*)d_in[13], d1Wi_f,  LL,     4 * HD, 4 * HD},
      {(const float*)d_in[14], d1Wh_f,  HD,     4 * HD, 4 * HD},
      {(const float*)d_in[16], d1Wi_b,  LL,     4 * HD, 4 * HD},
      {(const float*)d_in[17], d1Wh_b,  HD,     4 * HD, 4 * HD},
      {(const float*)d_in[19], d2Wi_f,  2 * HD, 4 * HD, 4 * HD},
      {(const float*)d_in[20], d2Wh_f,  HD,     4 * HD, 4 * HD},
      {(const float*)d_in[22], d2Wi_b,  2 * HD, 4 * HD, 4 * HD},
      {(const float*)d_in[23], d2Wh_b,  HD,     4 * HD, 4 * HD},
      {(const float*)d_in[25], Wout16,  2 * HD, VV,     VPAD},
  };
  for (const Job& jb : jobs) {
    long nel = (long)jb.K * jb.Npad;
    pack_b_f16<<<dim3((unsigned)((nel + 255) / 256)), 256, 0, stream>>>(
        jb.src, jb.dst, jb.K, jb.N, jb.Npad);
  }

  // ---- 2) embedding gather -> f16 ----
  {
    long n = (long)BB * TT * EE;
    embed_kernel<<<dim3((unsigned)((n + 255) / 256)), 256, 0, stream>>>(
        x, embed, e16, n);
  }

  const dim3 sgrid(BB / 16, 2);  // 32 blocks: both directions concurrent

  // ---- 3) encoder BiLSTM (E=128), outputs f32 [B,T,256] ----
  lstm_scan<EE, EE, false, false><<<sgrid, 256, 0, stream>>>(
      e16, (long)TT * EE, (long)EE,
      encWi_f, encWh_f, enc_b_f, encWi_b, encWh_b, enc_b_b,
      hs_enc, (long)TT * 2 * EE, (long)2 * EE, TT);

  // ---- 4) pool + mean/logvar + reparameterize ----
  pool_kernel<<<BB, 256, 0, stream>>>(x, hs_enc, W_mean, b_mean, W_logvar,
                                      b_logvar, eps, mean_out, logvar_out,
                                      z16, TT);

  // ---- 5) decoder layer 1 (Din=32, broadcast z; x@Wi hoisted) ----
  lstm_scan<LL, HD, true, true><<<sgrid, 256, 0, stream>>>(
      z16, (long)LL, 0L,
      d1Wi_f, d1Wh_f, d1_b_f, d1Wi_b, d1Wh_b, d1_b_b,
      d1o, (long)TT * 2 * HD, (long)2 * HD, TT);

  // ---- 6) decoder layer 2 (Din=512) ----
  lstm_scan<2 * HD, HD, false, true><<<sgrid, 256, 0, stream>>>(
      d1o, (long)TT * 2 * HD, (long)2 * HD,
      d2Wi_f, d2Wh_f, d2_b_f, d2Wi_b, d2Wh_b, d2_b_b,
      d2o, (long)TT * 2 * HD, (long)2 * HD, TT);

  // ---- 7) logits = d2o @ W_out + b_out  (M=64000, N=170->176, K=512) ----
  {
    const int M = BB * TT, K = 2 * HD;
    dim3 grid(M / 16, (VPAD / 16 + 7) / 8);  // (4000, 2)
    out_gemm<<<grid, 256, 0, stream>>>(d2o, Wout16, b_out, logits, M, VV,
                                       VPAD, K);
  }
}
